// SingleAttention_79637283602927
// MI455X (gfx1250) — matrix operand order repaired and checked by reading, not compiled, not run on hardware
//
#include <hip/hip_runtime.h>
#include <hip/hip_bf16.h>

// MI455X (gfx1250) single-head attention, f16 WMMA + f32 accumulation.
// (1) transpose/convert weights to f16, (2) WMMA projection GEMMs producing
// f16 q,k (row-major) and v^T ([B][DV][S]), (3) flash-attention with online
// softmax. Q staged via GLOBAL_LOAD_ASYNC_TO_LDS (ASYNCcnt); K/V blocks
// streamed by the Tensor Data Mover (TENSOR_LOAD_TO_LDS, TENSORcnt).

typedef _Float16 f16;
typedef __attribute__((ext_vector_type(16))) _Float16 v16h;
typedef __attribute__((ext_vector_type(8)))  _Float16 v8h;
typedef __attribute__((ext_vector_type(8)))  float    v8f;
typedef __attribute__((ext_vector_type(4)))  unsigned int u32x4;
typedef __attribute__((ext_vector_type(8)))  int          i32x8;
typedef __attribute__((ext_vector_type(4)))  int          i32x4;

constexpr int Bb = 8, Ss = 2048, Dd = 512, DKk = 512, DVv = 512;

// ---- async global -> LDS copy (16 bytes per lane), tracked by ASYNCcnt ----
__device__ __forceinline__ void async_cp16(const f16* gp, f16* lp) {
  // low 32 bits of the flat shared-aperture address = LDS byte offset
  unsigned loff = (unsigned)(unsigned long long)lp;
  asm volatile("global_load_async_to_lds_b128 %0, %1, off"
               :: "v"(loff), "v"(gp) : "memory");
}
__device__ __forceinline__ void async_wait0() {
  asm volatile("s_wait_asynccnt 0x0" ::: "memory");
}

// ---- Tensor Data Mover: 2D tile (f16 elements) global -> LDS -------------
// Descriptor packing per CDNA5 ISA ch.8 (D# group0 128b, group1 256b).
// tile_d1==0 -> 1D tile. Dims/strides in data_size (2-byte) units.
__device__ __forceinline__ void tdm_load_2d(const f16* gptr, f16* lptr,
                                            unsigned tile_d0, unsigned tile_d1,
                                            unsigned tensor_d0, unsigned tensor_d1,
                                            unsigned long long stride_d0) {
  const unsigned long long ga = (unsigned long long)gptr;    // byte address
  const unsigned lds = (unsigned)(unsigned long long)lptr;   // LDS byte offset
  u32x4 g0;
  g0[0] = 1u;                                   // count=1 valid user D#
  g0[1] = lds;                                  // lds_addr
  g0[2] = (unsigned)(ga & 0xffffffffu);         // global_addr[31:0]
  g0[3] = (unsigned)((ga >> 32) & 0x01ffffffu)  // global_addr[56:32]
        | (2u << 30);                           // type=2 ("image")
  i32x8 g1;
  g1[0] = (int)(1u << 16);                      // data_size=1 (2B), wg_mask=0
  g1[1] = (int)((tensor_d0 & 0xffffu) << 16);   // tensor_dim0[15:0]
  g1[2] = (int)((tensor_d0 >> 16) | ((tensor_d1 & 0xffffu) << 16));
  g1[3] = (int)((tensor_d1 >> 16) | (tile_d0 << 16));   // tile_dim0
  g1[4] = (int)(tile_d1 & 0xffffu);             // tile_dim1 (0=1D); tile_dim2=0
  g1[5] = (int)(stride_d0 & 0xffffffffu);       // tensor_dim0_stride[31:0]
  g1[6] = (int)((stride_d0 >> 32) & 0xffffu);   // stride[47:32]; dim1_stride=0
  g1[7] = 0;
  const i32x4 z4 = {0, 0, 0, 0};
#if __has_include(<hip/amd_detail/amd_gfx1250_TDM.h>)
  const i32x8 z8 = {0, 0, 0, 0, 0, 0, 0, 0};
  __builtin_amdgcn_tensor_load_to_lds(g0, g1, z4, z4, z8, 0);
#else
  __builtin_amdgcn_tensor_load_to_lds(g0, g1, z4, z4, 0);
#endif
}
__device__ __forceinline__ void tensor_wait0() {
#if __has_builtin(__builtin_amdgcn_s_wait_tensorcnt)
  __builtin_amdgcn_s_wait_tensorcnt(0);
#else
  asm volatile("s_wait_tensorcnt 0x0" ::: "memory");
#endif
}

// ---- fragment loads (per-lane) -------------------------------------------
// A fragment (16x32, 16-bit): lane<16 -> M=lane, K offsets {0..7},{16..23};
// lane>=16 -> M=lane-16, K offsets {8..15},{24..31}.
__device__ __forceinline__ v16h ld_a16(const f16* p) {
  v8h lo = *(const v8h*)(p);
  v8h hi = *(const v8h*)(p + 16);
  v16h a;
#pragma unroll
  for (int i = 0; i < 8; ++i) { a[i] = lo[i]; a[i + 8] = hi[i]; }
  return a;
}
// B fragment (32x16, 16-bit): lane<16 -> N=lane, K offsets {0..7},{16..23};
// lane>=16 -> N=lane-16, K offsets {8..15},{24..31}.
__device__ __forceinline__ v16h ld_b16(const f16* p) {
  v8h lo = *(const v8h*)(p);
  v8h hi = *(const v8h*)(p + 16);
  v16h b;
#pragma unroll
  for (int i = 0; i < 8; ++i) { b[i] = lo[i]; b[i + 8] = hi[i]; }
  return b;
}

__device__ __forceinline__ v8f wmma_f16(v16h a, v16h b, v8f c) {
  return __builtin_amdgcn_wmma_f32_16x16x32_f16(
      false, a, false, b, (short)0, c, false, false);
}

__device__ __forceinline__ float rmax16(float v) {
#pragma unroll
  for (int off = 8; off >= 1; off >>= 1)
    v = fmaxf(v, __shfl_xor(v, off, 32));
  return v;
}
__device__ __forceinline__ float rsum16(float v) {
#pragma unroll
  for (int off = 8; off >= 1; off >>= 1)
    v += __shfl_xor(v, off, 32);
  return v;
}

// ---- stage 1: weight transpose + f32->f16 convert ------------------------
__global__ __launch_bounds__(256) void prep_weights(
    const float* __restrict__ Wq, const float* __restrict__ Wk,
    const float* __restrict__ Wv, f16* __restrict__ WqT,
    f16* __restrict__ WkT, f16* __restrict__ WvT) {
  int i = blockIdx.x * 256 + threadIdx.x;   // 0 .. 512*512-1
  int n = i >> 9, d = i & 511;
  int src = d * 512 + n;                     // W is [D][N] row-major
  int dst = n * 512 + d;                     // WT is [N][D] row-major
  WqT[dst] = (f16)Wq[src];
  WkT[dst] = (f16)Wk[src];
  WvT[dst] = (f16)Wv[src];
}

// ---- stage 2: projection GEMM  out = X @ W + b ---------------------------
__global__ __launch_bounds__(128) void proj_kernel(
    const float* __restrict__ X, const f16* __restrict__ WT,
    const float* __restrict__ bias, f16* __restrict__ out, int transpose_out) {
  const int tid  = threadIdx.x;
  const int wave = tid >> 5, lane = tid & 31;
  const int half = lane >> 4, l16 = lane & 15;
  const int m0 = blockIdx.x * 64 + wave * 16;
  const int n0 = blockIdx.y * 128;

  v8f acc[8];
#pragma unroll
  for (int j = 0; j < 8; ++j)
#pragma unroll
    for (int r = 0; r < 8; ++r) acc[j][r] = 0.0f;

  const float* xrow = X + (size_t)(m0 + l16) * Dd;

  for (int kc = 0; kc < 16; ++kc) {          // K in chunks of 32
    const float* xp = xrow + kc * 32 + half * 8;
    v16h a;
#pragma unroll
    for (int i = 0; i < 8; ++i) {
      a[i]     = (f16)xp[i];
      a[i + 8] = (f16)xp[i + 16];
    }
#pragma unroll
    for (int j = 0; j < 8; ++j) {
      const f16* wp = WT + (size_t)(n0 + j * 16 + l16) * Dd + kc * 32 + half * 8;
      acc[j] = wmma_f16(a, ld_b16(wp), acc[j]);
    }
  }

  // bias + store.  C/D layout: comp r, lanes0-15: M=r, N=lane; lanes16-31: M=8+r.
#pragma unroll
  for (int j = 0; j < 8; ++j) {
    const int n = n0 + j * 16 + l16;
    const float bv = bias[n];
#pragma unroll
    for (int r = 0; r < 8; ++r) {
      const int rid = m0 + r + 8 * half;     // global row in [0, B*S)
      const float val = acc[j][r] + bv;
      if (!transpose_out) {
        out[(size_t)rid * DKk + n] = (f16)val;
      } else {
        const int bb = rid >> 11, ss = rid & 2047;   // S = 2048
        out[((size_t)bb * DVv + n) * Ss + ss] = (f16)val;
      }
    }
  }
}

// ---- stage 3: flash attention --------------------------------------------
// Grid (S/64, B), block 128 (4 waves). Wave owns 16 q-rows, full dv=512
// accumulator in registers (32 x v8f = 256 VGPRs; gfx1250 allows 1024).
// Q staged once (async-to-LDS); K/V blocks streamed by the TDM.
__global__ __launch_bounds__(128) void attn_kernel(
    const f16* __restrict__ Q, const f16* __restrict__ K,
    const f16* __restrict__ Vt, float* __restrict__ O) {
  extern __shared__ f16 smem[];
  f16* ldsK = smem;                        // [32 keys][512 d]   32KB
  f16* ldsV = smem + 32 * 512;             // [512 dv][32 kv]    32KB
  f16* ldsQ = smem + 2 * 32 * 512;         // [4 waves][16][512] 64KB
  const int tid  = threadIdx.x;
  const int wave = tid >> 5, lane = tid & 31;
  const int half = lane >> 4, l16 = lane & 15;
  f16* ldsP  = smem + 2 * 32 * 512 + 4 * 16 * 512 + wave * (16 * 32);  // 4KB
  f16* ldsQw = ldsQ + wave * (16 * 512);

  const int b  = blockIdx.y;
  const int m0 = blockIdx.x * 64 + wave * 16;        // q row within batch
  const f16* Qb = Q  + (size_t)b * Ss * DKk;
  const f16* Kb = K  + (size_t)b * Ss * DKk;
  const f16* Vb = Vt + (size_t)b * DVv * Ss;

  // stage this wave's 16x512 Q tile into LDS (async, wave-private)
  for (int i = lane; i < 1024; i += 32) {            // 1024 chunks of 8 f16
    const int mr = i >> 6, dco = (i & 63) << 3;
    async_cp16(Qb + (size_t)(m0 + mr) * DKk + dco, ldsQw + mr * 512 + dco);
  }
  async_wait0();

  v8f o[32];
#pragma unroll
  for (int t = 0; t < 32; ++t)
#pragma unroll
    for (int r = 0; r < 8; ++r) o[t][r] = 0.0f;

  float mrow[8], lrow[8];
#pragma unroll
  for (int r = 0; r < 8; ++r) { mrow[r] = -3.0e38f; lrow[r] = 0.0f; }

  const float sc = 0.022097086912079608f;  // 1/sqrt(S), S = 2048

  for (int j = 0; j < 64; ++j) {           // 64 blocks of 32 keys
    const int kv0 = j * 32;
    __syncthreads();                       // previous block fully consumed
    // TDM stage: K block = contiguous 32x512 f16 (1D, 32KB);
    //            V slice = 512 rows x 32 elems, row stride S (2D, 32KB).
    if (wave == 0) {
      tdm_load_2d(Kb + (size_t)kv0 * DKk, ldsK,
                  /*tile*/ 16384, 0, /*tensor*/ 16384, 1, /*stride*/ 16384);
      tdm_load_2d(Vb + kv0, ldsV,
                  /*tile*/ 32, 512, /*tensor*/ 2048, 512, /*stride*/ 2048);
      tensor_wait0();                      // issuing wave drains TENSORcnt
    }
    if (j + 1 < 64 && lane == 0) {
      __builtin_prefetch(Kb + (size_t)(kv0 + 32) * DKk, 0, 1);
      __builtin_prefetch(Vb + kv0 + 32, 0, 1);
    }
    __syncthreads();                       // publish LDS to all waves

    // scores: S[16x32] = Q[16x512] @ K^T[512x32], two 16x16 f32 tiles
    v8f s0, s1;
#pragma unroll
    for (int r = 0; r < 8; ++r) { s0[r] = 0.0f; s1[r] = 0.0f; }
#pragma unroll
    for (int kc = 0; kc < 16; ++kc) {
      v16h a  = ld_a16(ldsQw + l16 * 512 + kc * 32 + half * 8);
      v16h b0 = ld_b16(ldsK + l16 * 512 + kc * 32 + half * 8);
      v16h b1 = ld_b16(ldsK + (l16 + 16) * 512 + kc * 32 + half * 8);
      s0 = wmma_f16(a, b0, s0);
      s1 = wmma_f16(a, b1, s1);
    }

    // online softmax per row (row = r + 8*half in this wave's 16-row block)
    float p0[8], p1[8], cf[8];
#pragma unroll
    for (int r = 0; r < 8; ++r) {
      const float v0 = s0[r] * sc, v1 = s1[r] * sc;
      const float mx = rmax16(fmaxf(v0, v1));
      const float mn = fmaxf(mrow[r], mx);
      const float c  = __expf(mrow[r] - mn);
      const float e0 = __expf(v0 - mn);
      const float e1 = __expf(v1 - mn);
      const float rs = rsum16(e0 + e1);
      lrow[r] = lrow[r] * c + rs;
      mrow[r] = mn;
      cf[r] = c; p0[r] = e0; p1[r] = e1;
    }

    // rescale running output
#pragma unroll
    for (int t = 0; t < 32; ++t)
#pragma unroll
      for (int r = 0; r < 8; ++r) o[t][r] *= cf[r];

    // transpose P (C/D layout) -> A layout via wave-private LDS scratch
#pragma unroll
    for (int r = 0; r < 8; ++r) {
      const int mr = r + 8 * half;
      ldsP[mr * 32 + l16]      = (f16)p0[r];
      ldsP[mr * 32 + 16 + l16] = (f16)p1[r];
    }
    v16h pa = ld_a16(ldsP + l16 * 32 + half * 8);

    // O[16x512] += P[16x32] @ V[32x512]
#pragma unroll
    for (int t = 0; t < 32; ++t) {
      v16h vb = ld_b16(ldsV + (t * 16 + l16) * 32 + half * 8);
      o[t] = wmma_f16(pa, vb, o[t]);
    }
  }

  // epilogue: normalize and store fp32
  float inv[8];
#pragma unroll
  for (int r = 0; r < 8; ++r) inv[r] = 1.0f / lrow[r];
  float* Ob = O + (size_t)b * Ss * DVv;
#pragma unroll
  for (int t = 0; t < 32; ++t)
#pragma unroll
    for (int r = 0; r < 8; ++r) {
      const int m = m0 + r + 8 * half;
      Ob[(size_t)m * DVv + t * 16 + l16] = o[t][r] * inv[r];
    }
}

// ---- launcher -------------------------------------------------------------
extern "C" void kernel_launch(void* const* d_in, const int* in_sizes, int n_in,
                              void* d_out, int out_size, void* d_ws, size_t ws_size,
                              hipStream_t stream) {
  const float* x_q = (const float*)d_in[0];
  const float* x_k = (const float*)d_in[1];
  const float* x_v = (const float*)d_in[2];
  const float* Wq  = (const float*)d_in[3];
  const float* bq  = (const float*)d_in[4];
  const float* Wk  = (const float*)d_in[5];
  const float* bk  = (const float*)d_in[6];
  const float* Wv  = (const float*)d_in[7];
  const float* bv  = (const float*)d_in[8];
  float* out = (float*)d_out;

  // workspace layout (f16): q, k row-major; v transposed per batch; weightsT
  f16* qh  = (f16*)d_ws;
  f16* kh  = qh  + (size_t)Bb * Ss * DKk;
  f16* vTh = kh  + (size_t)Bb * Ss * DKk;
  f16* WqT = vTh + (size_t)Bb * DVv * Ss;
  f16* WkT = WqT + (size_t)Dd * DKk;
  f16* WvT = WkT + (size_t)Dd * DKk;

  prep_weights<<<dim3((512 * 512) / 256), dim3(256), 0, stream>>>(
      Wq, Wk, Wv, WqT, WkT, WvT);

  dim3 pg(256, 4);  // 256 M-tiles(64) x 4 N-tiles(128)
  proj_kernel<<<pg, dim3(128), 0, stream>>>(x_q, WqT, bq, qh, 0);
  proj_kernel<<<pg, dim3(128), 0, stream>>>(x_k, WkT, bk, kh, 0);
  proj_kernel<<<pg, dim3(128), 0, stream>>>(x_v, WvT, bv, vTh, 1);

  // LDS: K 32KB + V 32KB + Q 64KB + P 4KB = 132KB per workgroup
  const size_t shmem =
      (size_t)(2 * 32 * 512 + 4 * 16 * 512 + 4 * 16 * 32) * sizeof(f16);
  attn_kernel<<<dim3(Ss / 64, Bb), dim3(128), shmem, stream>>>(qh, kh, vTh, out);
}
